// DocumentGraphEncoder_85641647882648
// MI455X (gfx1250) — matrix operand order repaired and checked by reading, not compiled, not run on hardware
//
#include <hip/hip_runtime.h>

typedef __bf16 v16bf __attribute__((ext_vector_type(16)));
typedef float  v8f   __attribute__((ext_vector_type(8)));
typedef int    v4i   __attribute__((ext_vector_type(4)));
typedef __attribute__((address_space(1))) v4i gv4i;   // global int4
typedef __attribute__((address_space(3))) v4i lv4i;   // LDS int4

constexpr int kN = 30000, kE = 480000, kIN = 399, kHID = 256, kHEADS = 8, kG = 64;
constexpr int kMp = 30080;           // kN rounded up to 128 (== 235*128)
constexpr int kKpMax = 416;          // 399 rounded up to 32

#if defined(__HIP_DEVICE_COMPILE__) && defined(__has_builtin)
# if __has_builtin(__builtin_amdgcn_global_load_async_to_lds_b128) && \
     __has_builtin(__builtin_amdgcn_s_wait_asynccnt)
#  define HAVE_ASYNC_LDS 1
# endif
#endif
#ifndef HAVE_ASYNC_LDS
# define HAVE_ASYNC_LDS 0
#endif

// Address-space casts via integer round-trip (always-legal inttoptr):
// generic global VA == AS1 VA; generic shared VA low 32 bits == LDS offset
// (CDNA5 ISA 10.2).
__device__ __forceinline__ gv4i* as_global_v4i(const void* p) {
  return (gv4i*)(unsigned long long)(uintptr_t)p;
}
__device__ __forceinline__ lv4i* as_lds_v4i(void* p) {
  return (lv4i*)(unsigned)(uintptr_t)p;
}

__device__ __forceinline__ unsigned short f32_to_bf16_bits(float f) {
  unsigned u = __float_as_uint(f);
  u += 0x7FFFu + ((u >> 16) & 1u);          // round-to-nearest-even
  return (unsigned short)(u >> 16);
}
__device__ __forceinline__ unsigned fenc(float f) {   // order-preserving f32 -> u32
  unsigned u = __float_as_uint(f);
  return (u & 0x80000000u) ? ~u : (u | 0x80000000u);
}
__device__ __forceinline__ float fdec(unsigned e) {
  unsigned u = (e & 0x80000000u) ? (e & 0x7FFFFFFFu) : ~e;
  return __uint_as_float(u);
}
__device__ __forceinline__ float wave_sum(float v) {
  #pragma unroll
  for (int off = 16; off > 0; off >>= 1) v += __shfl_xor(v, off, 32);
  return v;
}
__device__ __forceinline__ void wait_async_0() {
#if HAVE_ASYNC_LDS
  __builtin_amdgcn_s_wait_asynccnt(0);
#endif
}

// ---------------------------------------------------------------------------
// bf16 pre-conversion (zero-padded) so GEMM tiles can be DMA'd guard-free.
// ---------------------------------------------------------------------------
__global__ __launch_bounds__(256) void cvt_pad_k(
    const float* __restrict__ in, unsigned short* __restrict__ out,
    int M, int K, int Kp, int total) {
  const int i = blockIdx.x * 256 + threadIdx.x;
  if (i >= total) return;
  const int r = i / Kp, k = i - r * Kp;
  out[i] = (r < M && k < K) ? f32_to_bf16_bits(in[(size_t)r * K + k]) : (unsigned short)0;
}
// weights [K][N] fp32 -> transposed padded [N][Kp] bf16
__global__ __launch_bounds__(256) void cvtT_k(
    const float* __restrict__ w, unsigned short* __restrict__ out,
    int K, int N, int Kp, int total) {
  const int i = blockIdx.x * 256 + threadIdx.x;
  if (i >= total) return;
  const int n = i / Kp, k = i - n * Kp;
  out[i] = (k < K) ? f32_to_bf16_bits(w[(size_t)k * N + n]) : (unsigned short)0;
}

// ---------------------------------------------------------------------------
// WMMA GEMM: C[M,Nout] = A[M,Kp]bf16 @ WT[Nout,Kp]bf16^T + bias (fp32 out)
// 256 threads = 8 waves; tile 128x64, BK=32; double-buffered LDS fed by
// async global->LDS DMA (overlapped with WMMA of the current tile).
// ---------------------------------------------------------------------------
#define BM 128
#define BN 64
#define BK 32

__device__ __forceinline__ void stage_tiles(
    const unsigned short* __restrict__ A, const unsigned short* __restrict__ WT,
    unsigned short* Abuf, unsigned short* Bbuf,
    int row0, int col0, int k0, int Kp, int tid)
{
#if HAVE_ASYNC_LDS
  #pragma unroll
  for (int q = 0; q < 2; ++q) {                       // A: 512 x 16B chunks
    const int id = tid + q * 256;
    const int rr = id >> 2, cc = id & 3;
    __builtin_amdgcn_global_load_async_to_lds_b128(
        as_global_v4i(A + (size_t)(row0 + rr) * Kp + (size_t)k0 + cc * 8),
        as_lds_v4i(Abuf + rr * BK + cc * 8), 0, 0);
  }
  {                                                    // B: 256 x 16B chunks
    const int n = tid >> 2, c = tid & 3;
    __builtin_amdgcn_global_load_async_to_lds_b128(
        as_global_v4i(WT + (size_t)(col0 + n) * Kp + (size_t)k0 + c * 8),
        as_lds_v4i(Bbuf + n * BK + c * 8), 0, 0);
  }
#else
  #pragma unroll
  for (int q = 0; q < 2; ++q) {
    const int id = tid + q * 256;
    const int rr = id >> 2, cc = id & 3;
    *reinterpret_cast<uint4*>(Abuf + rr * BK + cc * 8) =
        *reinterpret_cast<const uint4*>(A + (size_t)(row0 + rr) * Kp + (size_t)k0 + cc * 8);
  }
  {
    const int n = tid >> 2, c = tid & 3;
    *reinterpret_cast<uint4*>(Bbuf + n * BK + c * 8) =
        *reinterpret_cast<const uint4*>(WT + (size_t)(col0 + n) * Kp + (size_t)k0 + c * 8);
  }
#endif
}

__global__ __launch_bounds__(256) void gemm_wmma_bf16(
    const unsigned short* __restrict__ A,   // [Mp][Kp] bf16, zero padded
    const unsigned short* __restrict__ WT,  // [Nout][Kp] bf16, zero padded
    const float* __restrict__ bias, float* __restrict__ C,
    int M, int Kp, int Nout)
{
  __shared__ __align__(16) unsigned short Abuf[2][BM * BK];
  __shared__ __align__(16) unsigned short Bbuf[2][BN * BK];

  const int tid  = threadIdx.x;
  const int lane = tid & 31;
  const int wv   = tid >> 5;
  const int row0 = blockIdx.x * BM;
  const int col0 = blockIdx.y * BN;

  v8f acc[4] = {};
  const int ksteps = Kp / BK;

  stage_tiles(A, WT, Abuf[0], Bbuf[0], row0, col0, 0, Kp, tid);

  for (int ks = 0; ks < ksteps; ++ks) {
    wait_async_0();          // my DMA for tile ks complete
    __syncthreads();         // tile ks visible; all waves done reading buf[(ks+1)&1]
    if (ks + 1 < ksteps)     // overlap next tile's DMA with this tile's WMMAs
      stage_tiles(A, WT, Abuf[(ks + 1) & 1], Bbuf[(ks + 1) & 1],
                  row0, col0, (ks + 1) * BK, Kp, tid);

    const unsigned short* Ab = Abuf[ks & 1];
    const unsigned short* Bb = Bbuf[ks & 1];

    // A fragment: 16x32 bf16. lanes0-15: M=lane, K={0..7,16..23}; lanes16-31: K+=8
    union Frag { v16bf v; unsigned u[8]; };
    Frag af;
    {
      const int r = wv * 16 + (lane & 15);
      const unsigned* Ar = reinterpret_cast<const unsigned*>(Ab) + r * (BK / 2);
      const int h = (lane >> 4) << 2;   // 0 or 4 (uint pairs)
      af.u[0] = Ar[h + 0]; af.u[1] = Ar[h + 1];
      af.u[2] = Ar[h + 2]; af.u[3] = Ar[h + 3];
      af.u[4] = Ar[8 + h + 0]; af.u[5] = Ar[8 + h + 1];
      af.u[6] = Ar[8 + h + 2]; af.u[7] = Ar[8 + h + 3];
    }
    // B fragments: lanes0-15: N=lane, K={0..7,16..23}; lanes16-31: K+=8
    Frag bfr[4];
    {
      const int pb = (lane >> 4) << 2;  // 0 or 4 (uint pairs)
      #pragma unroll
      for (int t = 0; t < 4; ++t) {
        const int n = t * 16 + (lane & 15);
        const unsigned* Bn = reinterpret_cast<const unsigned*>(Bb) + n * (BK / 2);
        #pragma unroll
        for (int j = 0; j < 8; ++j) bfr[t].u[j] = Bn[pb + (j & 3) + ((j >> 2) << 3)];
      }
    }
    #pragma unroll
    for (int t = 0; t < 4; ++t) {
      acc[t] = __builtin_amdgcn_wmma_f32_16x16x32_bf16(
          false, af.v, false, bfr[t].v, (short)0, acc[t], false, false);
    }
  }

  // D layout: VGPR r -> M = r (lanes 0-15) / r+8 (lanes 16-31); N = lane&15
  const int rbase = row0 + wv * 16 + ((lane >> 4) << 3);
  #pragma unroll
  for (int t = 0; t < 4; ++t) {
    const int gcol = col0 + t * 16 + (lane & 15);
    const float bval = bias ? bias[gcol] : 0.f;
    #pragma unroll
    for (int r = 0; r < 8; ++r) {
      const int grow = rbase + r;
      if (grow < M) C[(long)grow * Nout + gcol] = acc[t][r] + bval;
    }
  }
}

// ---------------------------------------------------------------------------
// Elementwise helpers
// ---------------------------------------------------------------------------
__global__ __launch_bounds__(256) void fill_bias_k(
    float* __restrict__ out, const float* __restrict__ bias, int total, int Fmask) {
  const int i = blockIdx.x * 256 + threadIdx.x;
  if (i < total) out[i] = bias[i & Fmask];
}

__global__ __launch_bounds__(256) void elu_k(float* __restrict__ p, int total) {
  const int i = blockIdx.x * 256 + threadIdx.x;
  if (i < total) { const float v = p[i]; p[i] = v > 0.f ? v : __expf(v) - 1.f; }
}

// ---------------------------------------------------------------------------
// GATv2 edge phase. Wave per (edge, head); C channels strided by lane.
// ---------------------------------------------------------------------------
__global__ __launch_bounds__(256) void edge_logit_k(
    const int* __restrict__ src, const int* __restrict__ dst,
    const float* __restrict__ ea, const float* __restrict__ xl,
    const float* __restrict__ xr, const float* __restrict__ we,
    const float* __restrict__ att, float* __restrict__ logit,
    unsigned* __restrict__ segmax, int Etot, int H, int C)
{
  const int gw   = blockIdx.x * 8 + (threadIdx.x >> 5);
  const int lane = threadIdx.x & 31;
  if (gw >= Etot * H) return;
  const int e = gw / H, h = gw - e * H;
  const int s = src[e], d = dst[e];
  const int HC = H * C;
  const float a0 = ea[e * 4 + 0], a1 = ea[e * 4 + 1];
  const float a2 = ea[e * 4 + 2], a3 = ea[e * 4 + 3];
  float acc = 0.f;
  for (int c = lane; c < C; c += 32) {
    const int ch = h * C + c;
    const float ev = a0 * we[ch] + a1 * we[HC + ch] + a2 * we[2 * HC + ch] + a3 * we[3 * HC + ch];
    float sv = xl[(long)s * HC + ch] + xr[(long)d * HC + ch] + ev;
    sv = sv > 0.f ? sv : 0.2f * sv;                  // leaky relu
    acc += sv * att[ch];
  }
  acc = wave_sum(acc);
  if (lane == 0) {
    logit[(long)e * H + h] = acc;
    atomicMax(&segmax[(long)d * H + h], fenc(acc));
  }
}

__global__ __launch_bounds__(256) void edge_alpha_k(
    const int* __restrict__ dst, const float* __restrict__ logit,
    const unsigned* __restrict__ segmax, float* __restrict__ alpha,
    float* __restrict__ segden, int EH, int H)
{
  const int i = blockIdx.x * 256 + threadIdx.x;
  if (i >= EH) return;
  const int e = i / H, h = i - e * H;
  const int d = dst[e];
  const float a = __expf(logit[i] - fdec(segmax[(long)d * H + h]));
  alpha[i] = a;
  atomicAdd(&segden[(long)d * H + h], a);
}

__global__ __launch_bounds__(256) void edge_scatter_k(
    const int* __restrict__ src, const int* __restrict__ dst,
    const float* __restrict__ xl, const float* __restrict__ alpha,
    const float* __restrict__ segden, float* __restrict__ out,
    int Etot, int H, int C)
{
  const int gw   = blockIdx.x * 8 + (threadIdx.x >> 5);
  const int lane = threadIdx.x & 31;
  if (gw >= Etot * H) return;
  const int e = gw / H, h = gw - e * H;
  const int s = src[e], d = dst[e];
  const int HC = H * C;
  const float w = alpha[(long)e * H + h] / segden[(long)d * H + h];
  for (int c = lane; c < C; c += 32) {
    const int ch = h * C + c;
    atomicAdd(&out[(long)d * HC + ch], xl[(long)s * HC + ch] * w);
  }
}

// ---------------------------------------------------------------------------
// LayerNorm + gate logit, wave per node (256 = 8 values/lane)
// ---------------------------------------------------------------------------
__global__ __launch_bounds__(256) void ln_gate_k(
    float* __restrict__ h, const float* __restrict__ lnw, const float* __restrict__ lnb,
    const float* __restrict__ gw, const float* __restrict__ gb,
    float* __restrict__ gate, int Nn)
{
  const int wv = threadIdx.x >> 5, lane = threadIdx.x & 31;
  const int n = blockIdx.x * 8 + wv;
  if (n >= Nn) return;
  float v[8]; float s = 0.f;
  #pragma unroll
  for (int i = 0; i < 8; ++i) { v[i] = h[(long)n * 256 + i * 32 + lane]; s += v[i]; }
  s = wave_sum(s);
  const float mu = s * (1.f / 256.f);
  float q = 0.f;
  #pragma unroll
  for (int i = 0; i < 8; ++i) { const float d = v[i] - mu; q += d * d; }
  q = wave_sum(q);
  const float rs = rsqrtf(q * (1.f / 256.f) + 1e-5f);
  float g = 0.f;
  #pragma unroll
  for (int i = 0; i < 8; ++i) {
    const int ch = i * 32 + lane;
    const float y = (v[i] - mu) * rs * lnw[ch] + lnb[ch];
    h[(long)n * 256 + ch] = y;
    g += y * gw[ch];
  }
  g = wave_sum(g);
  if (lane == 0) gate[n] = g + gb[0];
}

// ---------------------------------------------------------------------------
// Group pooling: segment softmax over batch (G=64) then weighted sum
// ---------------------------------------------------------------------------
__global__ __launch_bounds__(256) void pool_max_k(
    const float* __restrict__ gate, const int* __restrict__ batch,
    unsigned* __restrict__ gmax, int Nn) {
  const int n = blockIdx.x * 256 + threadIdx.x;
  if (n < Nn) atomicMax(&gmax[batch[n]], fenc(gate[n]));
}

__global__ __launch_bounds__(256) void pool_exp_k(
    const float* __restrict__ gate, const int* __restrict__ batch,
    const unsigned* __restrict__ gmax, float* __restrict__ wexp,
    float* __restrict__ gden, int Nn) {
  const int n = blockIdx.x * 256 + threadIdx.x;
  if (n >= Nn) return;
  const int b = batch[n];
  const float w = __expf(gate[n] - fdec(gmax[b]));
  wexp[n] = w;
  atomicAdd(&gden[b], w);
}

__global__ __launch_bounds__(256) void pool_scatter_k(
    const float* __restrict__ values, const int* __restrict__ batch,
    const float* __restrict__ wexp, const float* __restrict__ gden,
    float* __restrict__ out, int Nn) {
  const int wv = threadIdx.x >> 5, lane = threadIdx.x & 31;
  const int n = blockIdx.x * 8 + wv;
  if (n >= Nn) return;
  const int b = batch[n];
  const float w = wexp[n] / gden[b];
  #pragma unroll
  for (int i = 0; i < 8; ++i) {
    const int ch = i * 32 + lane;
    atomicAdd(&out[(long)b * 256 + ch], values[(long)n * 256 + ch] * w);
  }
}

// ---------------------------------------------------------------------------
// Host orchestration
// ---------------------------------------------------------------------------
static void run_gemm(const unsigned short* A, const unsigned short* WT,
                     const float* bias, float* C,
                     int M, int Kp, int Nout, hipStream_t s) {
  dim3 g((M + BM - 1) / BM, Nout / BN);
  gemm_wmma_bf16<<<g, 256, 0, s>>>(A, WT, bias, C, M, Kp, Nout);
}

static void run_gat_layer(const float* hin, int Kin, int H, int C,
                          const float* wl, const float* bl, const float* wr, const float* br,
                          const float* we, const float* att, const float* bb,
                          const int* srcv, const int* dstv, const float* eattr,
                          unsigned short* abf, unsigned short* wtbf,
                          float* bufL, float* bufR, float* outb,
                          float* logit, float* alpha, unsigned* segmax, float* segden,
                          int do_elu, hipStream_t stream)
{
  const int HC = H * C;                       // always 256 here
  const int Kp = (Kin + 31) & ~31;
  const int atot = kMp * Kp, wtot = HC * Kp;
  cvt_pad_k<<<(atot + 255) / 256, 256, 0, stream>>>(hin, abf, kN, Kin, Kp, atot);
  cvtT_k<<<(wtot + 255) / 256, 256, 0, stream>>>(wl, wtbf, Kin, HC, Kp, wtot);
  run_gemm(abf, wtbf, bl, bufL, kN, Kp, HC, stream);
  cvtT_k<<<(wtot + 255) / 256, 256, 0, stream>>>(wr, wtbf, Kin, HC, Kp, wtot);
  run_gemm(abf, wtbf, br, bufR, kN, Kp, HC, stream);

  (void)hipMemsetAsync(segmax, 0, (size_t)kN * H * sizeof(unsigned), stream);
  (void)hipMemsetAsync(segden, 0, (size_t)kN * H * sizeof(float), stream);
  fill_bias_k<<<(kN * HC + 255) / 256, 256, 0, stream>>>(outb, bb, kN * HC, HC - 1);
  const int EH = kE * H;
  edge_logit_k<<<(EH + 7) / 8, 256, 0, stream>>>(srcv, dstv, eattr, bufL, bufR, we, att,
                                                 logit, segmax, kE, H, C);
  edge_alpha_k<<<(EH + 255) / 256, 256, 0, stream>>>(dstv, logit, segmax, alpha, segden, EH, H);
  edge_scatter_k<<<(EH + 7) / 8, 256, 0, stream>>>(srcv, dstv, bufL, alpha, segden, outb, kE, H, C);
  if (do_elu) elu_k<<<(kN * HC + 255) / 256, 256, 0, stream>>>(outb, kN * HC);
}

extern "C" void kernel_launch(void* const* d_in, const int* in_sizes, int n_in,
                              void* d_out, int out_size, void* d_ws, size_t ws_size,
                              hipStream_t stream) {
  (void)in_sizes; (void)n_in; (void)ws_size;
  const float* x     = (const float*)d_in[0];
  const int*   eidx  = (const int*)d_in[1];
  const int*   srcv  = eidx;
  const int*   dstv  = eidx + kE;
  const float* eattr = (const float*)d_in[2];
  const int*   batch = (const int*)d_in[3];
  const float* W[27];
  for (int i = 0; i < 27; ++i) W[i] = (const float*)d_in[4 + i];
  const float* ln_w = W[21], * ln_b = W[22];
  const float* gate_w = W[23], * gate_b = W[24];
  const float* tr_w = W[25], * tr_b = W[26];

  // workspace carve (fp32 words; all chunks 16B-aligned)
  float* p = (float*)d_ws;
  float* bufH = p;  p += (size_t)kN * kHID;
  float* bufL = p;  p += (size_t)kN * kHID;
  float* bufR = p;  p += (size_t)kN * kHID;
  float* logit = p; p += (size_t)kE * kHEADS;
  float* alpha = p; p += (size_t)kE * kHEADS;
  unsigned* segmax = (unsigned*)p; p += (size_t)kN * kHEADS;
  float* segden = p; p += (size_t)kN * kHEADS;
  float* gate = p;  p += kN;
  float* wexp = p;  p += kN;
  unsigned* gmax = (unsigned*)p; p += kG;
  float* gden = p;  p += kG;
  unsigned short* abf  = (unsigned short*)p; p += (size_t)kMp * kKpMax / 2;  // bf16 activations
  unsigned short* wtbf = (unsigned short*)p; p += (size_t)kHID * kKpMax / 2; // bf16 weights^T

  // layer 1: input x[N,399], H=8 C=32, ELU
  run_gat_layer(x, kIN, 8, 32, W[0], W[1], W[2], W[3], W[4], W[5], W[6],
                srcv, dstv, eattr, abf, wtbf, bufL, bufR, bufH,
                logit, alpha, segmax, segden, 1, stream);
  // layer 2: input bufH (consumed by cvt before fill_bias reuses it as output)
  run_gat_layer(bufH, kHID, 8, 32, W[7], W[8], W[9], W[10], W[11], W[12], W[13],
                srcv, dstv, eattr, abf, wtbf, bufL, bufR, bufH,
                logit, alpha, segmax, segden, 1, stream);
  // layer 3: single head, no ELU
  run_gat_layer(bufH, kHID, 1, 256, W[14], W[15], W[16], W[17], W[18], W[19], W[20],
                srcv, dstv, eattr, abf, wtbf, bufL, bufR, bufH,
                logit, alpha, segmax, segden, 0, stream);

  // LayerNorm (in place) + gate logits
  ln_gate_k<<<(kN + 7) / 8, 256, 0, stream>>>(bufH, ln_w, ln_b, gate_w, gate_b, gate, kN);

  // values = h_ln @ tr_w + tr_b -> bufL (via bf16 WMMA path)
  {
    const int Kp = kHID, atot = kMp * Kp, wtot = kHID * Kp;
    cvt_pad_k<<<(atot + 255) / 256, 256, 0, stream>>>(bufH, abf, kN, kHID, Kp, atot);
    cvtT_k<<<(wtot + 255) / 256, 256, 0, stream>>>(tr_w, wtbf, kHID, kHID, Kp, wtot);
    run_gemm(abf, wtbf, tr_b, bufL, kN, Kp, kHID, stream);
  }

  // pooling softmax over batch groups
  (void)hipMemsetAsync(gmax, 0, kG * sizeof(unsigned), stream);
  (void)hipMemsetAsync(gden, 0, kG * sizeof(float), stream);
  (void)hipMemsetAsync(d_out, 0, (size_t)out_size * sizeof(float), stream);
  pool_max_k<<<(kN + 255) / 256, 256, 0, stream>>>(gate, batch, gmax, kN);
  pool_exp_k<<<(kN + 255) / 256, 256, 0, stream>>>(gate, batch, gmax, wexp, gden, kN);
  pool_scatter_k<<<(kN + 7) / 8, 256, 0, stream>>>(bufL, batch, wexp, gden, (float*)d_out, kN);
}
